// TemplateSimpleNet_48206712930684
// MI455X (gfx1250) — matrix operand order repaired and checked by reading, not compiled, not run on hardware
//
#include <hip/hip_runtime.h>

// Problem constants (from the reference)
#define NN 160000      // nodes
#define FF 128         // features
#define EE 2560000     // edges
#define BB 8           // batch
#define PP 20000       // proteins per batch

typedef __bf16 v16bf __attribute__((ext_vector_type(16)));
typedef float  v8f   __attribute__((ext_vector_type(8)));

// ---------------------------------------------------------------------------
// init: agg = 0, deg = 1 (self loop), out = 0
// ---------------------------------------------------------------------------
__global__ void k_init(float* __restrict__ agg, float* __restrict__ deg,
                       float* __restrict__ out) {
    int i = blockIdx.x * 256 + threadIdx.x;
    if (i < NN * FF) agg[i] = 0.0f;
    if (i < NN)      deg[i] = 1.0f;
    if (i < BB * FF) out[i] = 0.0f;
}

// ---------------------------------------------------------------------------
// degree accumulation over dst, then rsqrt in place
// ---------------------------------------------------------------------------
__global__ void k_deg(const long long* __restrict__ ei, float* __restrict__ deg) {
    int e = blockIdx.x * 256 + threadIdx.x;
    if (e < EE) atomicAdd(&deg[(int)ei[(long long)EE + e]], 1.0f);
}

__global__ void k_rsqrt(float* __restrict__ deg) {
    int i = blockIdx.x * 256 + threadIdx.x;
    if (i < NN) deg[i] = rsqrtf(deg[i]);
}

// ---------------------------------------------------------------------------
// GEMM: xw = x @ W via v_wmma_f32_16x16x32_bf16
// Block = 256 threads = 8 waves; each wave -> 16 rows x 128 cols.
// W (128x128) converted to bf16 and stored in LDS pre-swizzled so a lane's
// B fragment (16 bf16) is one contiguous 32-byte read.
//
// B layout (16-bit, 32x16, K x N), per ISA 7.12.2:
//   lane = 16*h + n  (n = N index 0..15, h = K-half)
//   fragment element e (0..15) = W[32*c + 8*h + e (e<8) or 32*c + 16 + 8*h + e-8 (e>=8), 16*t + n]
// A layout (16-bit, 16x32): lane m = l&15, h = l>>4:
//   a[0..7]  = x[row, 32*c + 8*h + 0..7]
//   a[8..15] = x[row, 32*c + 16 + 8*h + 0..7]
// C/D (f32 16x16): vgpr r holds row (8*h + r), col n = lane&15
// ---------------------------------------------------------------------------
__global__ __launch_bounds__(256) void k_gemm(const float* __restrict__ x,
                                              const float* __restrict__ W,
                                              float* __restrict__ xw) {
    __shared__ __align__(32) __bf16 sW[128 * 128];

    const int tid = threadIdx.x;

    // Cooperative swizzled fill (coalesced reads of W over columns)
    for (int idx = tid; idx < 128 * 128; idx += 256) {
        int k    = idx >> 7;          // row of W (K dim)
        int ncol = idx & 127;         // col of W (N dim)
        int c    = k >> 5;            // K chunk (32 wide)
        int rem  = k & 31;
        int h    = (rem >> 3) & 1;    // lane half
        int e    = (rem & 7) | ((rem >> 4) << 3); // element within fragment
        int t    = ncol >> 4;         // N tile
        int n    = ncol & 15;
        int lane = (h << 4) | n;
        sW[((((t << 2) + c) * 32 + lane) << 4) + e] = (__bf16)W[idx];
    }
    __syncthreads();

    const int wave = tid >> 5;
    const int lane = tid & 31;
    const int m    = lane & 15;
    const int h    = lane >> 4;
    const int base = blockIdx.x * 128 + wave * 16;   // first row of this wave's tile
    const float* __restrict__ xr = x + (size_t)(base + m) * FF;

    v8f acc[8];
#pragma unroll
    for (int t = 0; t < 8; ++t)
        acc[t] = (v8f){0.f, 0.f, 0.f, 0.f, 0.f, 0.f, 0.f, 0.f};

#pragma unroll
    for (int c = 0; c < 4; ++c) {
        // A fragment: two contiguous 8-float runs, f32 -> bf16
        const float4 f0 = *(const float4*)(xr + 32 * c + 8 * h);
        const float4 f1 = *(const float4*)(xr + 32 * c + 8 * h + 4);
        const float4 f2 = *(const float4*)(xr + 32 * c + 16 + 8 * h);
        const float4 f3 = *(const float4*)(xr + 32 * c + 16 + 8 * h + 4);
        v16bf a;
        a[0]  = (__bf16)f0.x; a[1]  = (__bf16)f0.y; a[2]  = (__bf16)f0.z; a[3]  = (__bf16)f0.w;
        a[4]  = (__bf16)f1.x; a[5]  = (__bf16)f1.y; a[6]  = (__bf16)f1.z; a[7]  = (__bf16)f1.w;
        a[8]  = (__bf16)f2.x; a[9]  = (__bf16)f2.y; a[10] = (__bf16)f2.z; a[11] = (__bf16)f2.w;
        a[12] = (__bf16)f3.x; a[13] = (__bf16)f3.y; a[14] = (__bf16)f3.z; a[15] = (__bf16)f3.w;

#pragma unroll
        for (int t = 0; t < 8; ++t) {
            const v16bf b =
                *(const v16bf*)&sW[((((t << 2) + c) * 32 + lane) << 4)];
            acc[t] = __builtin_amdgcn_wmma_f32_16x16x32_bf16(
                /*neg_a=*/false, a, /*neg_b=*/false, b,
                /*c_mod=*/(short)0, acc[t],
                /*reuse_a=*/false, /*reuse_b=*/false);
        }
    }

    // Store D tiles: vgpr r -> row base + 8*h + r, col 16*t + (lane&15)
    const int n = lane & 15;
#pragma unroll
    for (int t = 0; t < 8; ++t) {
#pragma unroll
        for (int r = 0; r < 8; ++r) {
            xw[(size_t)(base + 8 * h + r) * FF + 16 * t + n] = acc[t][r];
        }
    }
}

// ---------------------------------------------------------------------------
// Edge scatter: agg[dst] += xw[src] * dinv[src]*dinv[dst]
// One wave per edge; each lane handles 4 features (float4 load + 4 atomics).
// ---------------------------------------------------------------------------
__global__ __launch_bounds__(256) void k_scatter(const float* __restrict__ xw,
                                                 const long long* __restrict__ ei,
                                                 const float* __restrict__ dinv,
                                                 float* __restrict__ agg) {
    const int wave = threadIdx.x >> 5;
    const int lane = threadIdx.x & 31;
    const long long e = (long long)blockIdx.x * 8 + wave;
    if (e >= EE) return;

    const int s = (int)ei[e];
    const int d = (int)ei[(long long)EE + e];
    const float norm = dinv[s] * dinv[d];

    const float4 v = *(const float4*)(xw + (size_t)s * FF + lane * 4);
    float* __restrict__ dp = agg + (size_t)d * FF + lane * 4;
    atomicAdd(dp + 0, v.x * norm);
    atomicAdd(dp + 1, v.y * norm);
    atomicAdd(dp + 2, v.z * norm);
    atomicAdd(dp + 3, v.w * norm);
}

// ---------------------------------------------------------------------------
// Fused self-loop + bias + masked pooling:
// out[b,f] += sum_p mask[b,p] * (agg[n,f] + xw[n,f]*dinv[n]^2 + bias[f])
// blockDim = 128 (one lane per feature), grid = (B, P/PCHUNK)
// ---------------------------------------------------------------------------
#define PCHUNK 250
__global__ __launch_bounds__(128) void k_pool(const float* __restrict__ xw,
                                              const float* __restrict__ agg,
                                              const float* __restrict__ dinv,
                                              const int* __restrict__ mask,
                                              const float* __restrict__ bias,
                                              float* __restrict__ out) {
    const int b = blockIdx.x;
    const int f = threadIdx.x;
    const int p0 = blockIdx.y * PCHUNK;

    float acc = 0.0f;
    float mcnt = 0.0f;
    for (int p = p0; p < p0 + PCHUNK; ++p) {
        if (mask[b * PP + p] != 0) {
            const size_t n = (size_t)b * PP + p;
            const float di = dinv[n];
            acc += agg[n * FF + f] + xw[n * FF + f] * (di * di);
            mcnt += 1.0f;
        }
    }
    acc += bias[f] * mcnt;
    atomicAdd(&out[b * FF + f], acc);
}

// ---------------------------------------------------------------------------
// launch
// ---------------------------------------------------------------------------
extern "C" void kernel_launch(void* const* d_in, const int* in_sizes, int n_in,
                              void* d_out, int out_size, void* d_ws, size_t ws_size,
                              hipStream_t stream) {
    const float*     x    = (const float*)d_in[0];
    const long long* ei   = (const long long*)d_in[1];  // int64 [2, E]
    const int*       mask = (const int*)d_in[2];        // int32 [B, P]
    const float*     W    = (const float*)d_in[3];      // [F, F]
    const float*     bias = (const float*)d_in[4];      // [F]
    float*           out  = (float*)d_out;              // [B, F]

    float* ws  = (float*)d_ws;
    float* xw  = ws;                             // N*F floats
    float* agg = ws + (size_t)NN * FF;           // N*F floats
    float* deg = ws + (size_t)2 * NN * FF;       // N floats (deg -> dinv in place)

    k_init<<<(NN * FF + 255) / 256, 256, 0, stream>>>(agg, deg, out);
    k_deg<<<(EE + 255) / 256, 256, 0, stream>>>(ei, deg);
    k_rsqrt<<<(NN + 255) / 256, 256, 0, stream>>>(deg);
    k_gemm<<<NN / 128, 256, 0, stream>>>(x, W, xw);
    k_scatter<<<EE / 8, 256, 0, stream>>>(xw, ei, deg, agg);
    dim3 gp(BB, PP / PCHUNK);
    k_pool<<<gp, 128, 0, stream>>>(xw, agg, deg, mask, bias, out);
}
